// KNNModule_31903017074734
// MI455X (gfx1250) — matrix operand order repaired and checked by reading, not compiled, not run on hardware
//
#include <hip/hip_runtime.h>

typedef __attribute__((ext_vector_type(16))) _Float16 v16h;
typedef __attribute__((ext_vector_type(8)))  _Float16 v8h;
typedef __attribute__((ext_vector_type(8)))  float    v8f;
typedef __attribute__((ext_vector_type(4)))  int      v4i;

#define BATCH 4
#define SEQ   4096
#define DIM   1024
#define KTOP  32
#define KSLAB 128          // K-depth staged per LDS slab
#define BPAD  136          // 128 + 8 halves pad -> 68-dword lane stride, no bank conflicts

#if __has_builtin(__builtin_amdgcn_global_load_async_to_lds_b128)
#define HAVE_ASYNC_LDS 1
#else
#define HAVE_ASYNC_LDS 0
#endif

#if __has_builtin(__builtin_amdgcn_s_wait_asynccnt)
#define WAIT_ASYNC(n) __builtin_amdgcn_s_wait_asynccnt(n)
#else
#define WAIT_ASYNC(n)
#endif

// ---------------------------------------------------------------------------
// Kernel 1: per-row L2 normalize, f32 -> f16 (WMMA operand precision).
// ---------------------------------------------------------------------------
__global__ __launch_bounds__(128) void knn_normalize(
    const float* __restrict__ emb, _Float16* __restrict__ out) {
  const int row = blockIdx.x;                      // BATCH*SEQ rows
  const float* src = emb + (size_t)row * DIM;
  _Float16*    dst = out + (size_t)row * DIM;

  float v[8];
  float ss = 0.f;
#pragma unroll
  for (int i = 0; i < 8; ++i) {
    v[i] = src[threadIdx.x + 128 * i];
    ss += v[i] * v[i];
  }
#pragma unroll
  for (int off = 16; off > 0; off >>= 1) ss += __shfl_xor(ss, off, 32);

  __shared__ float red[4];
  const int wv = threadIdx.x >> 5;
  if ((threadIdx.x & 31) == 0) red[wv] = ss;
  __syncthreads();
  const float tot = red[0] + red[1] + red[2] + red[3];
  const float inv = 1.0f / (sqrtf(tot) + 1e-8f);
#pragma unroll
  for (int i = 0; i < 8; ++i)
    dst[threadIdx.x + 128 * i] = (_Float16)(v[i] * inv);
}

// ---------------------------------------------------------------------------
// Stage one column's 128-K run (256 B) of the shared B slab into LDS.
// Async path: 16x GLOBAL_LOAD_ASYNC_TO_LDS_B128 (ASYNCcnt), else sync copy.
// ---------------------------------------------------------------------------
__device__ __forceinline__ void stage_slab(_Float16* lrow,
                                           const _Float16* grow) {
#if HAVE_ASYNC_LDS
#pragma unroll
  for (int i = 0; i < 16; ++i)
    __builtin_amdgcn_global_load_async_to_lds_b128(
        (v4i*)(grow + i * 8), (v4i*)(lrow + i * 8), 0, 0);
#else
#pragma unroll
  for (int i = 0; i < 16; ++i)
    *(v8h*)(lrow + i * 8) = *(const v8h*)(grow + i * 8);
#endif
}

// ---------------------------------------------------------------------------
// Kernel 2: fused A*A^T + streaming top-32.
// Block = 4 waves, each owns a 16-row strip; B slab shared via LDS with
// double-buffered async staging. 8 WMMA accumulators per wave (16x128 tile).
// ---------------------------------------------------------------------------
union Frag16 { v16h v; v8h h[2]; };

__global__ __launch_bounds__(128) void knn_topk(
    const _Float16* __restrict__ anorm, float* __restrict__ out) {
  __shared__ __align__(16) _Float16 s_B[2][128][BPAD];  // 2 x ~34 KB B slabs
  __shared__ float s_tile[4][16][16];   // score tile staging per wave
  __shared__ float s_ts[4][16][KTOP];   // per-row sorted top-k scores (desc)
  __shared__ int   s_ti[4][16][KTOP];   // per-row top-k indices

  const int tid  = threadIdx.x;
  const int lane = tid & 31;
  const int wv   = tid >> 5;
  const int tile = blockIdx.x * 4 + wv;        // 0..1023 row-strips
  const int b    = tile >> 8;                  // batch
  const int rowB = (tile & 255) * 16;          // strip base row in batch

  const _Float16* base = anorm + (size_t)b * SEQ * DIM;

  const int half = lane >> 4;                  // wave half (K-split)
  const int mn   = lane & 15;                  // row (A) / col (B) in tile

  if (lane < 16) {
#pragma unroll
    for (int j = 0; j < KTOP; ++j) {
      s_ts[wv][lane][j] = -3.0e38f;
      s_ti[wv][lane][j] = 0;
    }
  }

  // A row this lane feeds (16-bit A 16x32 layout: lanes 0-15 K 0-7/16-23,
  // lanes 16-31 K 8-15/24-31 -> two b128 loads at +half*8 and +16+half*8)
  const _Float16* aRow = base + (size_t)(rowB + mn) * DIM;

  // Preload first B slab (colB=0, k=0..127); each thread owns one column.
  stage_slab(&s_B[0][tid][0], base + (size_t)tid * DIM);
  int buf = 0;

  for (int cg = 0; cg < SEQ / 128; ++cg) {
    const int colB = cg * 128;
    v8f acc[8] = {};

    for (int s = 0; s < DIM / KSLAB; ++s) {
      // Prefetch next slab into the other buffer.
      const bool lastS  = (s == DIM / KSLAB - 1);
      const int  ncolB  = lastS ? colB + 128 : colB;
      const int  nk     = lastS ? 0 : (s + 1) * KSLAB;
      if (!(lastS && cg == SEQ / 128 - 1)) {
        stage_slab(&s_B[buf ^ 1][tid][0],
                   base + (size_t)(ncolB + tid) * DIM + nk);
        WAIT_ASYNC(16);   // in-order: <=16 outstanding => current slab done
      } else {
        WAIT_ASYNC(0);
      }
      __syncthreads();    // all waves' copies into s_B[buf] visible

      const _Float16* aR = aRow + s * KSLAB;
      for (int kk = 0; kk < KSLAB; kk += 32) {
        Frag16 a;
        a.h[0] = *(const v8h*)(aR + kk + half * 8);
        a.h[1] = *(const v8h*)(aR + kk + 16 + half * 8);
#pragma unroll
        for (int t = 0; t < 8; ++t) {
          // B 32x16 frag: lanes 0-15 K 0-7/16-23, lanes 16-31 K 8-15/24-31, col=mn.
          const _Float16* bP = &s_B[buf][t * 16 + mn][kk + half * 8];
          Frag16 bb;
          bb.h[0] = *(const v8h*)(bP);
          bb.h[1] = *(const v8h*)(bP + 16);
          acc[t] = __builtin_amdgcn_wmma_f32_16x16x32_f16(
              /*neg_a=*/false, a.v, /*neg_b=*/false, bb.v,
              /*c_mod=*/(short)0, acc[t], /*reuse_a=*/false, /*reuse_b=*/false);
        }
      }
      __syncthreads();    // everyone done reading s_B[buf] before reuse
      buf ^= 1;
    }

    // Streaming top-k update, one 16x16 subtile at a time.
#pragma unroll
    for (int t = 0; t < 8; ++t) {
      // C/D layout: VGPR v holds M = v + 8*half, N = lane&15.
#pragma unroll
      for (int v = 0; v < 8; ++v)
        s_tile[wv][v + 8 * half][mn] = acc[t][v];
      __builtin_amdgcn_wave_barrier();

      if (lane < 16) {
        const int r    = lane;
        const int grow = rowB + r;
        float* ts = s_ts[wv][r];
        int*   ti = s_ti[wv][r];
        for (int j = 0; j < 16; ++j) {
          const int c = colB + t * 16 + j;
          float sc = s_tile[wv][r][j];
          if (c == grow) sc -= 1.0e9f;         // mask diagonal
          if (sc > ts[KTOP - 1]) {             // threshold vs current min
            int p = KTOP - 1;
            while (p > 0 && ts[p - 1] < sc) {  // strict > keeps lower index
              ts[p] = ts[p - 1];
              ti[p] = ti[p - 1];
              --p;
            }
            ts[p] = sc;
            ti[p] = c;
          }
        }
      }
      __builtin_amdgcn_wave_barrier();
    }
  }

  // Emit: scores | indices(as float) | min_heap(first 16) | max_heap(-last 16)
  if (lane < 16) {
    const size_t row = (size_t)b * SEQ + rowB + lane;
    const size_t SCORES = (size_t)BATCH * SEQ * KTOP;       // 524288
    float* oS  = out + row * KTOP;
    float* oI  = out + SCORES + row * KTOP;
    float* oMn = out + 2 * SCORES + row * (KTOP / 2);
    float* oMx = out + 2 * SCORES + SCORES / 2 + row * (KTOP / 2);
    const float* ts = s_ts[wv][lane];
    const int*   ti = s_ti[wv][lane];
#pragma unroll
    for (int j = 0; j < KTOP; ++j) {
      oS[j] = ts[j];
      oI[j] = (float)ti[j];
    }
#pragma unroll
    for (int j = 0; j < KTOP / 2; ++j) {
      oMn[j] = ts[j];
      oMx[j] = -ts[KTOP / 2 + j];
    }
  }
}

// ---------------------------------------------------------------------------
extern "C" void kernel_launch(void* const* d_in, const int* in_sizes, int n_in,
                              void* d_out, int out_size, void* d_ws,
                              size_t ws_size, hipStream_t stream) {
  const float* emb = (const float*)d_in[0];
  float* out = (float*)d_out;
  _Float16* anorm = (_Float16*)d_ws;   // 4*4096*1024 f16 = 32 MB scratch

  knn_normalize<<<BATCH * SEQ, 128, 0, stream>>>(emb, anorm);
  knn_topk<<<(BATCH * SEQ / 16) / 4, 128, 0, stream>>>(anorm, out);
}
